// LSTM_3693671875218
// MI455X (gfx1250) — matrix operand order repaired and checked by reading, not compiled, not run on hardware
//
#include <hip/hip_runtime.h>

// ---------------- problem constants ----------------
constexpr int BATCH = 128;
constexpr int SEQ   = 1024;
constexpr int EMBD  = 100;
constexpr int HID   = 128;
constexpr int VOCAB = 10000;
constexpr int GATES = 4 * HID;   // 512
constexpr int KTOT  = 256;       // 128 (padded emb) + 128 (hid)

// ---------------- WMMA types ----------------
typedef __bf16 v16bf __attribute__((ext_vector_type(16)));
typedef __bf16 v8bf  __attribute__((ext_vector_type(8)));
typedef float  v8f   __attribute__((ext_vector_type(8)));

struct Bits32 { uint4 lo; uint4 hi; };

// A-matrix fragment (16x32 bf16): per lane (m = lane%16, kh = lane/16),
// elems are 16 consecutive bf16 at [kc*32 + kh*16].
__device__ inline v16bf load_a_frag(const __bf16* p) {
    Bits32 t;
    t.lo = *(const uint4*)(p);        // k .. k+7
    t.hi = *(const uint4*)(p + 8);    // k+8 .. k+15
    return __builtin_bit_cast(v16bf, t);
}

// B-matrix fragment (32x16 bf16): per lane (n = lane%16, kh = lane/16),
// elems are 16 consecutive bf16 at [kc*32 + kh*16] of column n's K stream.
__device__ inline v16bf load_b_frag(const __bf16* p) {
    Bits32 t;
    t.lo = *(const uint4*)(p);        // k .. k+7
    t.hi = *(const uint4*)(p + 8);    // k+8 .. k+15
    return __builtin_bit_cast(v16bf, t);
}

// native v_rcp_f32 (~1 ulp) instead of the IEEE divide expansion:
// removes ~8 VALU ops per divide from the serialized recurrence path.
__device__ inline float fast_rcp(float x) { return __builtin_amdgcn_rcpf(x); }

__device__ inline float sigmoid_f(float x) {
    return fast_rcp(1.0f + __expf(-x));
}
__device__ inline float tanh_f(float x) {
    x = fminf(fmaxf(x, -15.0f), 15.0f);
    float e = __expf(2.0f * x);
    return (e - 1.0f) * fast_rcp(e + 1.0f);
}

// ---------------- prep: W_cat = [W_ih(pad 100->128) | W_hh] in bf16 ----------------
__global__ void prep_wcat(const float* __restrict__ W_ih,
                          const float* __restrict__ W_hh,
                          __bf16* __restrict__ Wcat) {
    int idx = blockIdx.x * 256 + threadIdx.x;          // [0, 512*256)
    if (idx >= GATES * KTOT) return;
    int g = idx >> 8;          // gate row 0..511
    int k = idx & 255;         // fused-K 0..255
    float v;
    if (k < 128) v = (k < EMBD) ? W_ih[g * EMBD + k] : 0.0f;
    else         v = W_hh[g * HID + (k - 128)];
    Wcat[idx] = (__bf16)v;
}

__global__ void prep_wout(const float* __restrict__ W_out,
                          __bf16* __restrict__ Wb) {
    int idx = blockIdx.x * 256 + threadIdx.x;          // [0, 10000*128)
    if (idx >= VOCAB * HID) return;
    Wb[idx] = (__bf16)W_out[idx];
}

// ---------------- persistent fused LSTM kernel ----------------
// grid = 8 blocks (16 batch rows each), block = 256 threads (8 waves).
// Wave w owns gate tiles {w, w+8, w+16, w+24} = i/f/g/o for hid [16w, 16w+16).
// All 32 weight B-fragments (256 VGPRs) and all 8 A-fragments (64 VGPRs) are
// register-resident so the per-step critical path is: 16 ds_load_b128 in
// flight -> progressive waits -> 32 WMMAs -> native-rcp gate math.
__global__ __launch_bounds__(256) void lstm_kernel(
        const int*   __restrict__ x,       // [B, T]
        const float* __restrict__ emb,     // [VOCAB, EMB]
        const __bf16* __restrict__ Wcat_g, // [512, 256]
        __bf16* __restrict__ h_out)        // [B, HID] bf16
{
    __shared__ __align__(16) __bf16 act[16 * KTOT];      // 8 KB: [e_pad | h]

    const int tid  = threadIdx.x;
    const int bg   = blockIdx.x;       // batch tile
    const int lane = tid & 31;
    const int w    = tid >> 5;         // wave id 0..7
    const int n    = lane & 15;        // N col / A row (lane%16)
    const int kh   = lane >> 4;        // lane half

    // ---- hoist all loop-invariant weight fragments into VGPRs ----
    v16bf bw[4][8];
    {
        const __bf16* wr0 = Wcat_g + (size_t)((w      ) * 16 + n) * KTOT; // i
        const __bf16* wr1 = Wcat_g + (size_t)((w +  8) * 16 + n) * KTOT;  // f
        const __bf16* wr2 = Wcat_g + (size_t)((w + 16) * 16 + n) * KTOT;  // g
        const __bf16* wr3 = Wcat_g + (size_t)((w + 24) * 16 + n) * KTOT;  // o
        #pragma unroll
        for (int kc = 0; kc < 8; ++kc) {
            bw[0][kc] = load_b_frag(wr0 + kc * 32 + kh * 16);
            bw[1][kc] = load_b_frag(wr1 + kc * 32 + kh * 16);
            bw[2][kc] = load_b_frag(wr2 + kc * 32 + kh * 16);
            bw[3][kc] = load_b_frag(wr3 + kc * 32 + kh * 16);
        }
    }

    // zero h-region of activation tile (h0 = 0)
    for (int i = tid; i < 16 * HID; i += 256)
        act[(i >> 7) * KTOT + 128 + (i & 127)] = (__bf16)0.0f;

    float cst[8];                      // cell state, (m = r + 8*kh, hid = 16w+n)
    float hv[8];
    #pragma unroll
    for (int r = 0; r < 8; ++r) { cst[r] = 0.0f; hv[r] = 0.0f; }

    // embedding-gather role: thread -> (em = tid/16, ek = 8*(tid%16))
    const int em = tid >> 4;
    const int ek = (tid & 15) * 8;
    const long tokBase = (long)(bg * 16 + em) * SEQ;

    const __bf16* arow = &act[(size_t)n * KTOT];
    __bf16* epub = &act[em * KTOT + ek];

    // ---- prologue: gather e(0) into registers ----
    float ereg[8];
    {
        int tok = x[tokBase];
        const float* er = emb + (long)tok * EMBD;
        #pragma unroll
        for (int j = 0; j < 8; ++j) {
            int k = ek + j;
            ereg[j] = (k < EMBD) ? er[k] : 0.0f;
        }
    }

    for (int t = 0; t < SEQ; ++t) {
        // ---- publish e(t): one packed 16B LDS store ----
        {
            v8bf ev;
            #pragma unroll
            for (int j = 0; j < 8; ++j) ev[j] = (__bf16)ereg[j];
            *(v8bf*)epub = ev;
        }
        __syncthreads();   // e(t) and h(t-1) ready in act

        // ---- prefetch e(t+1); overlaps with WMMA + gate-math phase ----
        float enext[8];
        {
            int tn = (t + 1 < SEQ) ? (t + 1) : (SEQ - 1);
            int tok = x[tokBase + tn];
            const float* er = emb + (long)tok * EMBD;
            #pragma unroll
            for (int j = 0; j < 8; ++j) {
                int k = ek + j;
                enext[j] = (k < EMBD) ? er[k] : 0.0f;
            }
        }

        // ---- preload ALL 8 A-fragments (16 ds_load_b128 in flight) ----
        v16bf af[8];
        #pragma unroll
        for (int kc = 0; kc < 8; ++kc)
            af[kc] = load_a_frag(arow + kc * 32 + kh * 16);

        // ---- gates[16,512]: 32 WMMAs, 4 independent accumulator chains ----
        v8f acc0 = {}, acc1 = {}, acc2 = {}, acc3 = {};
        #pragma unroll
        for (int kc = 0; kc < 8; ++kc) {
            acc0 = __builtin_amdgcn_wmma_f32_16x16x32_bf16(false, af[kc], false, bw[0][kc], (short)0, acc0, false, false);
            acc1 = __builtin_amdgcn_wmma_f32_16x16x32_bf16(false, af[kc], false, bw[1][kc], (short)0, acc1, false, false);
            acc2 = __builtin_amdgcn_wmma_f32_16x16x32_bf16(false, af[kc], false, bw[2][kc], (short)0, acc2, false, false);
            acc3 = __builtin_amdgcn_wmma_f32_16x16x32_bf16(false, af[kc], false, bw[3][kc], (short)0, acc3, false, false);
        }

        // ---- elementwise LSTM cell update, all in registers ----
        #pragma unroll
        for (int r = 0; r < 8; ++r) {
            float ig = sigmoid_f(acc0[r]);
            float fg = sigmoid_f(acc1[r]);
            float gg = tanh_f(acc2[r]);
            float og = sigmoid_f(acc3[r]);
            float c  = fg * cst[r] + ig * gg;
            cst[r]   = c;
            hv[r]    = og * tanh_f(c);
        }
        __syncthreads();   // all WMMA reads of act done before overwrite

        // ---- publish h(t) into act h-region ----
        #pragma unroll
        for (int r = 0; r < 8; ++r) {
            int m = (kh << 3) + r;
            act[m * KTOT + 128 + w * 16 + n] = (__bf16)hv[r];
        }

        #pragma unroll
        for (int j = 0; j < 8; ++j) ereg[j] = enext[j];
    }

    // ---- final h -> global (bf16, row-major [B, HID]) ----
    #pragma unroll
    for (int r = 0; r < 8; ++r) {
        int m = (kh << 3) + r;
        h_out[(size_t)(bg * 16 + m) * HID + w * 16 + n] = (__bf16)hv[r];
    }
}

// ---------------- output head: logits = h @ W_out^T + b ----------------
// grid = (625 vocab tiles, 8 batch tiles), block = 1 wave (32 threads).
__global__ __launch_bounds__(32) void out_gemm(
        const __bf16* __restrict__ hbf,   // [B, HID]
        const __bf16* __restrict__ Wob,   // [VOCAB, HID]
        const float*  __restrict__ b_out, // [VOCAB]
        float* __restrict__ logits)       // [B, VOCAB]
{
    const int vt = blockIdx.x;        // vocab tile 0..624
    const int bt = blockIdx.y;        // batch tile 0..7
    const int lane = threadIdx.x;
    const int n  = lane & 15;
    const int kh = lane >> 4;

    const __bf16* arow = hbf + (size_t)(bt * 16 + n) * HID;   // A: m = lane%16
    const __bf16* brow = Wob + (size_t)(vt * 16 + n) * HID;   // B: col n = lane%16

    v8f acc = {};
    #pragma unroll
    for (int kc = 0; kc < 4; ++kc) {   // K = 128 = 4 x 32
        v16bf a = load_a_frag(arow + kc * 32 + kh * 16);
        v16bf b = load_b_frag(brow + kc * 32 + kh * 16);
        acc = __builtin_amdgcn_wmma_f32_16x16x32_bf16(false, a, false, b, (short)0, acc, false, false);
    }
    float bias = b_out[vt * 16 + n];
    #pragma unroll
    for (int r = 0; r < 8; ++r) {
        int m = (kh << 3) + r;
        logits[(size_t)(bt * 16 + m) * VOCAB + vt * 16 + n] = acc[r] + bias;
    }
}

// ---------------- softmax over VOCAB per batch row ----------------
__global__ __launch_bounds__(256) void softmax_k(
        const float* __restrict__ logits, float* __restrict__ out)
{
    __shared__ float red[256];
    const int row = blockIdx.x;
    const int tid = threadIdx.x;
    const float* lr = logits + (size_t)row * VOCAB;

    float mx = -3.402823e38f;
    for (int i = tid; i < VOCAB; i += 256) mx = fmaxf(mx, lr[i]);
    red[tid] = mx; __syncthreads();
    for (int s = 128; s > 0; s >>= 1) {
        if (tid < s) red[tid] = fmaxf(red[tid], red[tid + s]);
        __syncthreads();
    }
    mx = red[0]; __syncthreads();

    float sum = 0.0f;
    for (int i = tid; i < VOCAB; i += 256) sum += __expf(lr[i] - mx);
    red[tid] = sum; __syncthreads();
    for (int s = 128; s > 0; s >>= 1) {
        if (tid < s) red[tid] += red[tid + s];
        __syncthreads();
    }
    float inv = 1.0f / red[0];   // once per row; keep IEEE divide
    for (int i = tid; i < VOCAB; i += 256)
        out[(size_t)row * VOCAB + i] = __expf(lr[i] - mx) * inv;
}

// ---------------- launcher ----------------
extern "C" void kernel_launch(void* const* d_in, const int* in_sizes, int n_in,
                              void* d_out, int out_size, void* d_ws, size_t ws_size,
                              hipStream_t stream) {
    const int*   x     = (const int*)  d_in[0];
    const float* emb   = (const float*)d_in[1];
    const float* W_ih  = (const float*)d_in[2];
    const float* W_hh  = (const float*)d_in[3];
    const float* W_out = (const float*)d_in[4];
    const float* b_out = (const float*)d_in[5];
    float* out = (float*)d_out;

    // workspace layout (all 16B-aligned)
    char* ws = (char*)d_ws;
    __bf16* Wcat  = (__bf16*)(ws);                                   //   262,144 B
    __bf16* Woutb = (__bf16*)(ws + 262144);                          // 2,560,000 B
    __bf16* hbf   = (__bf16*)(ws + 262144 + 2560000);                //    32,768 B
    float*  logit = (float*) (ws + 262144 + 2560000 + 32768);        // 5,120,000 B

    prep_wcat<<<(GATES * KTOT + 255) / 256, 256, 0, stream>>>(W_ih, W_hh, Wcat);
    prep_wout<<<(VOCAB * HID + 255) / 256, 256, 0, stream>>>(W_out, Woutb);

    lstm_kernel<<<BATCH / 16, 256, 0, stream>>>(x, emb, Wcat, hbf);

    out_gemm<<<dim3(VOCAB / 16, BATCH / 16), 32, 0, stream>>>(hbf, Woutb, b_out, logit);
    softmax_k<<<BATCH, 256, 0, stream>>>(logit, out);
}
